// SphericalFourierNeuralOperatorNet_46763603919534
// MI455X (gfx1250) — compile-verified, matrix operand order checked by reading, not run on hardware
//
#include <hip/hip_runtime.h>

// ---------------------------------------------------------------------------
// SFNO forward for MI455X (gfx1250, wave32, WMMA).
// All math is f16-operand / f32-accumulate GEMMs through one generic batched
// WMMA kernel (v_wmma_f32_16x16x32_f16) with double-buffered async
// global->LDS staging (ASYNCcnt), plus small convert/reorder kernels.
// ---------------------------------------------------------------------------

typedef __attribute__((ext_vector_type(16))) _Float16 v16h;
typedef __attribute__((ext_vector_type(8)))  float    v8f;
typedef __attribute__((ext_vector_type(4)))  int      v4i;

#define FLAG_ACC  1
#define FLAG_RELU 2
#define FLAG_BIAS 4

#if defined(__HIP_DEVICE_COMPILE__) && \
    __has_builtin(__builtin_amdgcn_global_load_async_to_lds_b128) && \
    __has_builtin(__builtin_amdgcn_s_wait_asynccnt)
#define USE_ASYNC_LDS 1
#else
#define USE_ASYNC_LDS 0
#endif

// ---------------- problem constants ----------------
static constexpr long NLAT = 128, NLON = 256, LMAX = 128, MMAX = 128;
static constexpr long P    = NLAT * NLON;   // 32768 pixels
static constexpr long EMB  = 256, HID = 512, IN_CH = 36, OUT_CH = 6;
static constexpr int  N_LAYERS = 4;

// ---------------- workspace layout (bytes) ----------------
static constexpr size_t OFF_HA   = 0;                        // h (pixel-major f32 [P,256]) -- also residual
static constexpr size_t OFF_HB   = OFF_HA + 33554432ull;     // h2 (pixel-major f32 [P,256])
static constexpr size_t OFF_MP   = OFF_HB + 33554432ull;     // MLP / decoder hidden f32 [P,512]
static constexpr size_t OFF_A4   = OFF_MP + 67108864ull;     // hCM_f16 / u2CM_f32 / res_f16 / m_f16 / d_f16
static constexpr size_t OFF_A5   = OFF_A4 + 33554432ull;     // Fre+Fim f32 / f2re+f2im f32 / h2_f16 / cat_f16 / xT_f16
static constexpr size_t OFF_A6   = OFF_A5 + 33554432ull;     // FreT+FimT f16 / bre+bim f16 / outP f32
static constexpr size_t OFF_A7   = OFF_A6 + 16777216ull;     // are+aim f32 / bre+bim f32
static constexpr size_t OFF_A8   = OFF_A7 + 33554432ull;     // acat1+acat2 f16 / f2re+f2im f16
static constexpr size_t OFF_A9   = OFF_A8 + 33554432ull;     // specTcat f16 [l][o][512]
static constexpr size_t OFF_WT   = OFF_A9 + 33554432ull;     // weights + tables (f16)

static constexpr size_t OFF_WENC = OFF_WT;                   // [256,64]
static constexpr size_t OFF_SKW  = OFF_WENC + 32768ull;      // 4 x [256,256]
static constexpr size_t OFF_W1   = OFF_SKW  + 524288ull;     // 4 x [512,256]
static constexpr size_t OFF_W2   = OFF_W1   + 1048576ull;    // 4 x [256,512]
static constexpr size_t OFF_DW1  = OFF_W2   + 1048576ull;    // [512,320]
static constexpr size_t OFF_DW2  = OFF_DW1  + 327680ull;     // [64,512] (rows>=6 zero)
static constexpr size_t OFF_CCOS = OFF_DW2  + 65536ull;      // [128,256]
static constexpr size_t OFF_CSIN = OFF_CCOS + 65536ull;      // [128,256]
static constexpr size_t OFF_ICOS = OFF_CSIN + 65536ull;      // [256,128]
static constexpr size_t OFF_ISIN = OFF_ICOS + 65536ull;      // [256,128]
static constexpr size_t OFF_PW   = OFF_ISIN + 65536ull;      // [m][l][h] f16 (pct*wq)
static constexpr size_t OFF_PI   = OFF_PW   + 4194304ull;    // [m][h][l] f16 (pct)

// ---------------------------------------------------------------------------
// helpers
// ---------------------------------------------------------------------------
struct H32 { uint4 a, b; };

// 16-byte global -> LDS copy, async when the CDNA5 builtin is available.
// Builtin prototype (probed): (global v4i*, local v4i*, imm offset, imm cpol).
__device__ __forceinline__ void cp16_g2l(_Float16* l, const _Float16* g) {
#if USE_ASYNC_LDS
  __builtin_amdgcn_global_load_async_to_lds_b128(
      (__attribute__((address_space(1))) v4i*)(void*)g,
      (__attribute__((address_space(3))) v4i*)(void*)l, 0, 0);
#else
  *reinterpret_cast<uint4*>(l) = *reinterpret_cast<const uint4*>(g);
#endif
}
__device__ __forceinline__ void wait_async_le3() {
#if USE_ASYNC_LDS
  __builtin_amdgcn_s_wait_asynccnt(3);   // 3 async ops per wave per staged tile
#endif
}
__device__ __forceinline__ void wait_async_le0() {
#if USE_ASYNC_LDS
  __builtin_amdgcn_s_wait_asynccnt(0);
#endif
}

// 16x32 f16 A/B fragment from an LDS row (row stride applied by caller).
// ISA 7.12.2: lanes 0-15 hold K {0..7,16..23}; lanes 16-31 hold K {8..15,24..31}.
__device__ __forceinline__ v16h load_frag_lds(const _Float16* __restrict__ row, int half) {
  const _Float16* p = row + half * 8;
  H32 t;
  t.a = *reinterpret_cast<const uint4*>(p);
  t.b = *reinterpret_cast<const uint4*>(p + 16);
  return __builtin_bit_cast(v16h, t);
}

// ---------------------------------------------------------------------------
// Generic batched GEMM:  D[r,n] (+)= sum_k A[r,k] * BT[n,k]  (+bias, relu)
// A: [M,K] f16 row-major (lda, batch stride sA)
// BT:[N,K] f16 row-major (ldb, batch stride sB)
// D: f32, element (r,n) at D + z*sD + r*sdr + n*sdn
// Requirements: M % 128 == 0 handled by grid (all M are 128/256/32768),
// N % 32 == 0, K % 32 == 0, BT has at least ceil64(N) valid (padded) rows.
// Block = 8 waves = 128x64 tile; wave tile 32x32 (4 WMMA accum frags).
// K pipeline: double-buffered async global->LDS staging.
// ---------------------------------------------------------------------------
__global__ void __launch_bounds__(256)
k_gemm(const _Float16* __restrict__ A, long lda, long sA,
       const _Float16* __restrict__ B, long ldb, long sB,
       float* __restrict__ D, long sdr, long sdn, long sD,
       const float* __restrict__ bias,
       int M, int N, int K, int flags)
{
  constexpr int LDS_LD = 40;                       // padded row stride in halves (80B, 16B-aligned)
  __shared__ __align__(16) _Float16 sAt[2][128 * LDS_LD];
  __shared__ __align__(16) _Float16 sBt[2][64 * LDS_LD];

  const long z = blockIdx.z;
  A += z * sA;  B += z * sB;  D += z * sD;

  const int tid  = threadIdx.x;
  const int lane = tid & 31;
  const int wave = tid >> 5;
  const long mblk = (long)blockIdx.y * 128;
  const long nblk = (long)blockIdx.x * 64;
  const int wm = (wave & 3) * 32;                  // wave row offset in block tile
  const int wn = (wave >> 2) * 32;                 // wave col offset in block tile
  const int r = lane & 15, half = lane >> 4;

  // stage one K=32 tile (A: 128 rows, B: 64 rows) into LDS buffer b
  auto stage = [&](int b, long k) {
    int c0 = tid, c1 = tid + 256;                  // A: 512 chunks of 16B, 2/thread
    int row0 = c0 >> 2, q0 = (c0 & 3) * 8;
    int row1 = c1 >> 2, q1 = (c1 & 3) * 8;
    cp16_g2l(&sAt[b][row0 * LDS_LD + q0], A + (mblk + row0) * lda + k + q0);
    cp16_g2l(&sAt[b][row1 * LDS_LD + q1], A + (mblk + row1) * lda + k + q1);
    int rowB = tid >> 2, qB = (tid & 3) * 8;       // B: 256 chunks, 1/thread
    cp16_g2l(&sBt[b][rowB * LDS_LD + qB], B + (nblk + rowB) * ldb + k + qB);
  };

  v8f c00 = {}, c01 = {}, c10 = {}, c11 = {};

  const int kT = K >> 5;
  stage(0, 0);
  for (int i = 0; i < kT; ++i) {
    if (i) __syncthreads();                        // WAR: buffer (i+1)&1 free to refill
    if (i + 1 < kT) { stage((i + 1) & 1, (long)(i + 1) * 32); wait_async_le3(); }
    else           { wait_async_le0(); }
    __syncthreads();                               // tile i visible to all waves

    const _Float16* a  = &sAt[i & 1][0];
    const _Float16* bt = &sBt[i & 1][0];
    v16h a0 = load_frag_lds(a  + (wm + r)      * LDS_LD, half);
    v16h a1 = load_frag_lds(a  + (wm + 16 + r) * LDS_LD, half);
    v16h b0 = load_frag_lds(bt + (wn + r)      * LDS_LD, half);
    v16h b1 = load_frag_lds(bt + (wn + 16 + r) * LDS_LD, half);
    c00 = __builtin_amdgcn_wmma_f32_16x16x32_f16(false, a0, false, b0, (short)0, c00, false, false);
    c01 = __builtin_amdgcn_wmma_f32_16x16x32_f16(false, a0, false, b1, (short)0, c01, false, false);
    c10 = __builtin_amdgcn_wmma_f32_16x16x32_f16(false, a1, false, b0, (short)0, c10, false, false);
    c11 = __builtin_amdgcn_wmma_f32_16x16x32_f16(false, a1, false, b1, (short)0, c11, false, false);
  }

  const long m0 = mblk + wm, n0 = nblk + wn;
  if (n0 >= N || m0 >= M) return;                  // after all barriers -- safe

  // D layout (ISA): lanes 0-15 -> n = lane, rows = v (VGPR idx); lanes 16-31 -> n = lane-16, rows = 8+v.
  const long nA = n0 + r, nB = n0 + 16 + r;
  const float bvA = (flags & FLAG_BIAS) ? bias[nA] : 0.f;
  const float bvB = (flags & FLAG_BIAS) ? bias[nB] : 0.f;
#pragma unroll
  for (int v = 0; v < 8; ++v) {
    const long mA = m0 + (long)half * 8 + v;
    const long mB = mA + 16;
    {
      long idx = mA * sdr + nA * sdn; float val = c00[v] + bvA;
      if (flags & FLAG_ACC) val += D[idx];
      if (flags & FLAG_RELU) val = fmaxf(val, 0.f);
      D[idx] = val;
    }
    {
      long idx = mA * sdr + nB * sdn; float val = c01[v] + bvB;
      if (flags & FLAG_ACC) val += D[idx];
      if (flags & FLAG_RELU) val = fmaxf(val, 0.f);
      D[idx] = val;
    }
    {
      long idx = mB * sdr + nA * sdn; float val = c10[v] + bvA;
      if (flags & FLAG_ACC) val += D[idx];
      if (flags & FLAG_RELU) val = fmaxf(val, 0.f);
      D[idx] = val;
    }
    {
      long idx = mB * sdr + nB * sdn; float val = c11[v] + bvB;
      if (flags & FLAG_ACC) val += D[idx];
      if (flags & FLAG_RELU) val = fmaxf(val, 0.f);
      D[idx] = val;
    }
  }
}

// ---------------------------------------------------------------------------
// Generic f32 -> f16 convert with 3-D index remap and zero padding.
// dst[(i0*D1 + i1)*D2 + i2] = (i0<V0 && i2<D2v) ? src[i0*s0+i1*s1+i2*s2] : 0
// ---------------------------------------------------------------------------
__global__ void k_cvt(_Float16* __restrict__ dst, const float* __restrict__ src,
                      long D0, long D1, long D2, long D2v,
                      long s0, long s1, long s2, long V0)
{
  long t = (long)blockIdx.x * blockDim.x + threadIdx.x;
  long total = D0 * D1 * D2;
  if (t >= total) return;
  long i2 = t % D2;
  long rr = t / D2;
  long i1 = rr % D1;
  long i0 = rr / D1;
  float v = 0.f;
  if (i0 < V0 && i2 < D2v) v = src[i0 * s0 + i1 * s1 + i2 * s2];
  dst[t] = (_Float16)v;
}

// DFT / inverse-DFT tables (f16).  Forward: F[m] = sum_n u[n] e^{-2pi i mn/N}.
// Inverse (real, MMAX=128 coeffs, coeff 128 zero):
//   u[n] = sum_m s_m (Re F cos - Im F sin),  s_0 = 1/N, s_m = 2/N.
__global__ void k_tables(_Float16* __restrict__ ccos, _Float16* __restrict__ csin,
                         _Float16* __restrict__ icos, _Float16* __restrict__ isin)
{
  int t = blockIdx.x * blockDim.x + threadIdx.x;
  if (t >= 32768) return;
  int m = t >> 8, n = t & 255;                 // ccos/csin: [m][n]
  double th = 6.283185307179586476925287 * (double)(m * n) / 256.0;
  ccos[t] = (_Float16)cos(th);
  csin[t] = (_Float16)(-sin(th));
  int n2 = t >> 7, m2 = t & 127;               // icos/isin: [n][m]
  double th2 = 6.283185307179586476925287 * (double)(m2 * n2) / 256.0;
  double sm = (m2 == 0) ? (1.0 / 256.0) : (2.0 / 256.0);
  icos[t] = (_Float16)(sm * cos(th2));
  isin[t] = (_Float16)(-(sm * sin(th2)));
}

// pw[m][l][h] = pct[l][m][h]*wq[h] (SHT A-operand);  pi[m][h][l] = pct[l][m][h] (ISHT A-operand)
__global__ void k_pwpi(_Float16* __restrict__ pw, _Float16* __restrict__ pi,
                       const float* __restrict__ pct, const float* __restrict__ wq)
{
  long t = (long)blockIdx.x * blockDim.x + threadIdx.x;
  if (t >= 128l * 128 * 128) return;
  long h = t & 127, l = (t >> 7) & 127, m = t >> 14;
  float p = pct[(l * 128 + m) * 128 + h];
  pw[t] = (_Float16)(p * wq[h]);               // t == (m*128+l)*128+h
  pi[(m * 128 + h) * 128 + l] = (_Float16)p;
}

// specTcat[l][o][0:256]=spec_re[o][i][l], [256:512]=spec_im[o][i][l]
__global__ void k_speccat(_Float16* __restrict__ dst,
                          const float* __restrict__ re, const float* __restrict__ im)
{
  long t = (long)blockIdx.x * blockDim.x + threadIdx.x;
  if (t >= 128l * 256 * 512) return;
  long i = t % 512, o = (t / 512) % 256, l = t / (512 * 256);
  float v = (i < 256) ? re[(o * 256 + i) * 128 + l]
                      : im[(o * 256 + (i - 256)) * 128 + l];
  dst[t] = (_Float16)v;
}

// Complex-GEMM operand concat: acat1 = [are | -aim], acat2 = [aim | are],
// both [m][l][512] from are/aim f32 [m][l][256].
__global__ void k_acat(_Float16* __restrict__ a1, _Float16* __restrict__ a2,
                       const float* __restrict__ are, const float* __restrict__ aim)
{
  long t = (long)blockIdx.x * blockDim.x + threadIdx.x;
  if (t >= 128l * 128 * 512) return;
  long i = t % 512, rl = t / 512;
  if (i < 256) {
    a1[t] = (_Float16)are[rl * 256 + i];
    a2[t] = (_Float16)aim[rl * 256 + i];
  } else {
    long ii = i - 256;
    a1[t] = (_Float16)(-aim[rl * 256 + ii]);
    a2[t] = (_Float16)are[rl * 256 + ii];
  }
}

// h[p][c] += pos_embed[c][p]
__global__ void k_posadd(float* __restrict__ h, const float* __restrict__ pos)
{
  long t = (long)blockIdx.x * blockDim.x + threadIdx.x;
  if (t >= P * EMB) return;
  long c = t & 255, p = t >> 8;
  h[t] += pos[c * P + p];
}

// h2[p][c] = u2CM[c][p]   (transpose CM f32 -> PM f32)
__global__ void k_addT(float* __restrict__ h2, const float* __restrict__ u2)
{
  long t = (long)blockIdx.x * blockDim.x + threadIdx.x;
  if (t >= P * EMB) return;
  long c = t & 255, p = t >> 8;
  h2[t] = u2[c * P + p];
}

// cat[p][c] = c<256 ? h[p][c] : (c<292 ? x[c-256][p] : 0)   (f16, K padded to 320)
__global__ void k_cat(_Float16* __restrict__ cat, const float* __restrict__ h,
                      const float* __restrict__ x)
{
  long t = (long)blockIdx.x * blockDim.x + threadIdx.x;
  if (t >= P * 320) return;
  long c = t % 320, p = t / 320;
  float v = 0.f;
  if (c < 256)       v = h[p * 256 + c];
  else if (c < 292)  v = x[(c - 256) * P + p];
  cat[t] = (_Float16)v;
}

// out[c][p] = outP[p][c]  (c < 6)
__global__ void k_out(float* __restrict__ out, const float* __restrict__ outP)
{
  long t = (long)blockIdx.x * blockDim.x + threadIdx.x;
  if (t >= OUT_CH * P) return;
  long c = t / P, p = t % P;
  out[t] = outP[p * 32 + c];
}

// ---------------------------------------------------------------------------
// host-side launch
// ---------------------------------------------------------------------------
extern "C" void kernel_launch(void* const* d_in, const int* in_sizes, int n_in,
                              void* d_out, int out_size, void* d_ws, size_t ws_size,
                              hipStream_t stream)
{
  (void)in_sizes; (void)n_in; (void)out_size; (void)ws_size;

  const float* x      = (const float*)d_in[0];
  const float* pct    = (const float*)d_in[1];
  const float* wq     = (const float*)d_in[2];
  const float* pos    = (const float*)d_in[3];
  const float* encw   = (const float*)d_in[4];
  const float* specre = (const float*)d_in[5];
  const float* specim = (const float*)d_in[6];
  const float* skw    = (const float*)d_in[7];
  const float* skb    = (const float*)d_in[8];
  const float* w1     = (const float*)d_in[9];
  const float* b1     = (const float*)d_in[10];
  const float* w2     = (const float*)d_in[11];
  const float* b2     = (const float*)d_in[12];
  const float* dw1    = (const float*)d_in[13];
  const float* db1    = (const float*)d_in[14];
  const float* dw2    = (const float*)d_in[15];

  char* ws = (char*)d_ws;
#define F32(off) ((float*)(ws + (off)))
#define H16(off) ((_Float16*)(ws + (off)))

  auto gemm = [&](size_t aOff, long lda, long sA, size_t bOff, long ldb, long sB,
                  size_t dOff, long sdr, long sdn, long sD, const float* bias,
                  int M, int N, int K, int Z, int flags) {
    dim3 grid((unsigned)((N + 63) / 64), (unsigned)((M + 127) / 128), (unsigned)Z);
    k_gemm<<<grid, 256, 0, stream>>>(H16(aOff), lda, sA, H16(bOff), ldb, sB,
                                     F32(dOff), sdr, sdn, sD, bias, M, N, K, flags);
  };
  auto cvt = [&](size_t dstOff, const float* src, long D0, long D1, long D2, long D2v,
                 long s0, long s1, long s2, long V0) {
    long total = D0 * D1 * D2;
    k_cvt<<<(unsigned)((total + 255) / 256), 256, 0, stream>>>(H16(dstOff), src,
        D0, D1, D2, D2v, s0, s1, s2, V0);
  };

  // ---- one-time tables & weight conversion ----
  k_tables<<<128, 256, 0, stream>>>(H16(OFF_CCOS), H16(OFF_CSIN), H16(OFF_ICOS), H16(OFF_ISIN));
  k_pwpi<<<(unsigned)((2097152 + 255) / 256), 256, 0, stream>>>(H16(OFF_PW), H16(OFF_PI), pct, wq);

  cvt(OFF_WENC, encw, 256, 1, 64, 36, 36, 0, 1, 256);             // [256,36]->[256,64]
  for (int l = 0; l < N_LAYERS; ++l) {
    cvt(OFF_SKW + (size_t)l * 131072, skw + (size_t)l * 65536,  256, 1, 256, 256, 256, 0, 1, 256);
    cvt(OFF_W1  + (size_t)l * 262144, w1  + (size_t)l * 131072, 512, 1, 256, 256, 256, 0, 1, 512);
    cvt(OFF_W2  + (size_t)l * 262144, w2  + (size_t)l * 131072, 256, 1, 512, 512, 512, 0, 1, 256);
  }
  cvt(OFF_DW1, dw1, 512, 1, 320, 292, 292, 0, 1, 512);            // [512,292]->[512,320]
  cvt(OFF_DW2, dw2, 64, 1, 512, 512, 512, 0, 1, 6);               // [6,512]->[64,512] rows>=6 zero

  // ---- encoder:  hA[p][c] = enc_w @ x + pos ----
  cvt(OFF_A5, x, P, 1, 64, 36, 1, 0, P, P);                        // xT f16 [P,64]
  gemm(OFF_A5, 64, 0, OFF_WENC, 64, 0, OFF_HA, 256, 1, 0, nullptr,
       (int)P, 256, 64, 1, 0);
  k_posadd<<<(unsigned)((P * EMB + 255) / 256), 256, 0, stream>>>(F32(OFF_HA), pos);

  const size_t FRE = OFF_A5, FIM = OFF_A5 + 16777216ull;           // f32 [32768,128]
  const size_t FRT = OFF_A6, FIT = OFF_A6 + 8388608ull;            // f16 [m][c][h]
  const size_t ARE = OFF_A7, AIM = OFF_A7 + 16777216ull;           // f32 [m][l][256]
  const size_t AC1 = OFF_A8, AC2 = OFF_A8 + 16777216ull;           // f16 [m][l][512]
  const size_t BRE = OFF_A7, BIM = OFF_A7 + 16777216ull;           // f32 [m][o][l] (reuse)
  const size_t BRH = OFF_A6, BIH = OFF_A6 + 8388608ull;            // f16 (reuse)
  const size_t F2R = OFF_A5, F2I = OFF_A5 + 16777216ull;           // f32 [c][h][m] (reuse)
  const size_t F2RH = OFF_A8, F2IH = OFF_A8 + 8388608ull;          // f16 (reuse)

  for (int l = 0; l < N_LAYERS; ++l) {
    // ---- SHT: DFT over longitude as GEMM ----
    cvt(OFF_A4, F32(OFF_HA), 256, 128, 256, 256, 1, 65536, 256, 256);    // hCM f16 [(c,h)][w]
    gemm(OFF_A4, 256, 0, OFF_CCOS, 256, 0, FRE, 128, 1, 0, nullptr, (int)P, 128, 256, 1, 0);
    gemm(OFF_A4, 256, 0, OFF_CSIN, 256, 0, FIM, 128, 1, 0, nullptr, (int)P, 128, 256, 1, 0);
    // reorder to per-m operands [m][c][h]
    cvt(FRT, F32(FRE), 128, 256, 128, 128, 1, 16384, 128, 128);
    cvt(FIT, F32(FIM), 128, 256, 128, 128, 1, 16384, 128, 128);
    // Legendre contraction, batched over m:  a[m][l][c]
    gemm(OFF_PW, 128, 16384, FRT, 128, 32768, ARE, 256, 1, 32768, nullptr, 128, 256, 128, 128, 0);
    gemm(OFF_PW, 128, 16384, FIT, 128, 32768, AIM, 256, 1, 32768, nullptr, 128, 256, 128, 128, 0);

    // ---- spectral complex block-diagonal multiply, batched over l ----
    k_acat<<<(unsigned)((8388608 + 255) / 256), 256, 0, stream>>>(H16(AC1), H16(AC2), F32(ARE), F32(AIM));
    k_speccat<<<(unsigned)((16777216 + 255) / 256), 256, 0, stream>>>(
        H16(OFF_A9), specre + (size_t)l * 8388608, specim + (size_t)l * 8388608);
    // D element (o,m) for batch l stored at bre + m*32768 + o*128 + l  -> layout [m][o][l]
    gemm(OFF_A9, 512, 131072, AC1, 65536, 512, BRE, 128, 32768, 1, nullptr, 256, 128, 512, 128, 0);
    gemm(OFF_A9, 512, 131072, AC2, 65536, 512, BIM, 128, 32768, 1, nullptr, 256, 128, 512, 128, 0);

    // ---- ISHT: Legendre synthesis (batched over m) then inverse DFT ----
    cvt(BRH, F32(BRE), 32768, 1, 128, 128, 128, 0, 1, 32768);
    cvt(BIH, F32(BIM), 32768, 1, 128, 128, 128, 0, 1, 32768);
    // D element (h,c) for batch m at f2 + c*16384 + h*128 + m  -> layout [(c,h)][m]
    gemm(OFF_PI, 128, 16384, BRH, 128, 32768, F2R, 128, 16384, 1, nullptr, 128, 256, 128, 128, 0);
    gemm(OFF_PI, 128, 16384, BIH, 128, 32768, F2I, 128, 16384, 1, nullptr, 128, 256, 128, 128, 0);
    cvt(F2RH, F32(F2R), 32768, 1, 128, 128, 128, 0, 1, 32768);
    cvt(F2IH, F32(F2I), 32768, 1, 128, 128, 128, 0, 1, 32768);
    // u2CM [(c,h)][w] = f2re @ icos + f2im @ isin
    gemm(F2RH, 128, 0, OFF_ICOS, 128, 0, OFF_A4, 256, 1, 0, nullptr, (int)P, 256, 128, 1, 0);
    gemm(F2IH, 128, 0, OFF_ISIN, 128, 0, OFF_A4, 256, 1, 0, nullptr, (int)P, 256, 128, 1, FLAG_ACC);

    // ---- h2 = isht + skip_w @ res + skip_b ----
    k_addT<<<(unsigned)((P * EMB + 255) / 256), 256, 0, stream>>>(F32(OFF_HB), F32(OFF_A4));
    cvt(OFF_A4, F32(OFF_HA), P, 1, 256, 256, 256, 0, 1, P);              // res f16
    gemm(OFF_A4, 256, 0, OFF_SKW + (size_t)l * 131072, 256, 0, OFF_HB, 256, 1, 0,
         skb + (size_t)l * 256, (int)P, 256, 256, 1, FLAG_ACC | FLAG_BIAS);

    // ---- MLP + outer residual (accumulates into hA which still holds res) ----
    cvt(OFF_A5, F32(OFF_HB), P, 1, 256, 256, 256, 0, 1, P);              // h2 f16
    gemm(OFF_A5, 256, 0, OFF_W1 + (size_t)l * 262144, 256, 0, OFF_MP, 512, 1, 0,
         b1 + (size_t)l * 512, (int)P, 512, 256, 1, FLAG_BIAS | FLAG_RELU);
    cvt(OFF_A4, F32(OFF_MP), P, 1, 512, 512, 512, 0, 1, P);              // m f16
    gemm(OFF_A4, 512, 0, OFF_W2 + (size_t)l * 262144, 512, 0, OFF_HA, 256, 1, 0,
         b2 + (size_t)l * 256, (int)P, 256, 512, 1, FLAG_ACC | FLAG_BIAS);
  }

  // ---- decoder ----
  k_cat<<<(unsigned)((P * 320 + 255) / 256), 256, 0, stream>>>(H16(OFF_A5), F32(OFF_HA), x);
  gemm(OFF_A5, 320, 0, OFF_DW1, 320, 0, OFF_MP, 512, 1, 0, db1,
       (int)P, 512, 320, 1, FLAG_BIAS | FLAG_RELU);
  cvt(OFF_A4, F32(OFF_MP), P, 1, 512, 512, 512, 0, 1, P);
  gemm(OFF_A4, 512, 0, OFF_DW2, 512, 0, OFF_A6, 32, 1, 0, nullptr,
       (int)P, 32, 512, 1, 0);
  k_out<<<(unsigned)((OUT_CH * P + 255) / 256), 256, 0, stream>>>((float*)d_out, F32(OFF_A6));

#undef F32
#undef H16
}
